// Non_local_MH_56865366999105
// MI455X (gfx1250) — matrix operand order repaired and checked by reading, not compiled, not run on hardware
//
#include <hip/hip_runtime.h>

#define NB   16
#define CCH  512
#define DPIX 1024
#define PIX  (NB*DPIX)   // 16384
#define NHD  8
#define HDIM 64
#define GRP  128
#define EPSV 1e-5f

typedef __bf16 bf16_t;
typedef __attribute__((ext_vector_type(16))) bf16_t v16bf;
typedef __attribute__((ext_vector_type(8)))  bf16_t bf16x8;
typedef __attribute__((ext_vector_type(8)))  float  v8f;

__device__ __forceinline__ v8f wmma_bf16(v16bf a, v16bf b, v8f c) {
  return __builtin_amdgcn_wmma_f32_16x16x32_bf16(false, a, false, b, (short)0, c, false, false);
}

// A fragment: 16x32 bf16 tile, row-major with row stride ld.
// Lane (h = lane>>4, m = lane&15): elems 0..7 -> K = h*8+e, elems 8..15 -> K = 16+h*8+(e-8)
__device__ __forceinline__ v16bf load_a(const bf16_t* tile, int ld) {
  int lane = threadIdx.x & 31;
  int m = lane & 15, h = lane >> 4;
  const bf16_t* r = tile + (size_t)m * ld;
  bf16x8 c0 = *(const bf16x8*)(r + h * 8);
  bf16x8 c1 = *(const bf16x8*)(r + 16 + h * 8);
  v16bf a;
#pragma unroll
  for (int i = 0; i < 8; ++i) { a[i] = c0[i]; a[i + 8] = c1[i]; }
  return a;
}

// B fragment from B-transposed storage: Bt is N x K row-major (row stride ld), B[k][n] = Bt[n][k].
// Lane (h, n = lane&15): elems 0..7 -> K = h*8+e, elems 8..15 -> K = 16+h*8+(e-8)
__device__ __forceinline__ v16bf load_bt(const bf16_t* tileT, int ld) {
  int lane = threadIdx.x & 31;
  int n = lane & 15, h = lane >> 4;
  const bf16_t* r = tileT + (size_t)n * ld;
  bf16x8 c0 = *(const bf16x8*)(r + h * 8);
  bf16x8 c1 = *(const bf16x8*)(r + 16 + h * 8);
  v16bf b;
#pragma unroll
  for (int i = 0; i < 8; ++i) { b[i] = c0[i]; b[i + 8] = c1[i]; }
  return b;
}

// ---------------- kernel 0: pack weights to bf16, zero GN stats ----------------
__global__ void pack_weights(const float* Wq, const float* Wk, const float* Wv, const float* Wo,
                             bf16_t* out, float* stats) {
  int tid = blockIdx.x * blockDim.x + threadIdx.x;
  int m = tid >> 18;            // 512*512 = 2^18
  int e = tid & 262143;
  const float* s = (m == 0) ? Wq : (m == 1) ? Wk : (m == 2) ? Wv : Wo;
  out[tid] = (bf16_t)s[e];
  if (tid < 2 * NB * GRP) stats[tid] = 0.0f;   // 4096 floats: sums + sumsqs
}

// ---------------- kernel 1: X [N][C][d] f32 -> Xt [P][C] bf16 (LDS tiled transpose) ----------------
__global__ void transpose_x(const float* __restrict__ X, bf16_t* __restrict__ Xt) {
  __shared__ float t[16][17];
  int n = blockIdx.z, c0 = blockIdx.y * 16, hw0 = blockIdx.x * 16;
  int tx = threadIdx.x, ty = threadIdx.y;
  t[ty][tx] = X[(size_t)n * CCH * DPIX + (size_t)(c0 + ty) * DPIX + hw0 + tx];
  __syncthreads();
  Xt[(size_t)(n * DPIX + hw0 + ty) * CCH + c0 + tx] = (bf16_t)t[tx][ty];
}

// ---------------- kernel 2: QKV GEMM.  z=0: Wq->K (ch-major), z=1: Wk->Q (pix-major), z=2: Wv->V ----------------
__global__ __launch_bounds__(32) void qkv_gemm(const bf16_t* __restrict__ Wb,
                                               const float* bq, const float* bk, const float* bv,
                                               const bf16_t* __restrict__ Xt,
                                               bf16_t* __restrict__ Kcm, bf16_t* __restrict__ Qt,
                                               bf16_t* __restrict__ Vcm) {
  int z = blockIdx.z;
  const bf16_t* A = Wb + (size_t)z * CCH * CCH;
  const float* bias = (z == 0) ? bq : (z == 1) ? bk : bv;
  int m0 = blockIdx.y * 64, p0 = blockIdx.x * 64;
  int lane = threadIdx.x & 31, j = lane & 15, h = lane >> 4;

  v8f acc[4][4];
#pragma unroll
  for (int tm = 0; tm < 4; ++tm) {
    v8f ini;
#pragma unroll
    for (int r = 0; r < 8; ++r) ini[r] = bias[m0 + tm * 16 + h * 8 + r];
#pragma unroll
    for (int tn = 0; tn < 4; ++tn) acc[tm][tn] = ini;
  }

  for (int kk = 0; kk < CCH; kk += 32) {
    v16bf af[4], bfr[4];
#pragma unroll
    for (int tm = 0; tm < 4; ++tm) af[tm] = load_a(A + (size_t)(m0 + tm * 16) * CCH + kk, CCH);
#pragma unroll
    for (int tn = 0; tn < 4; ++tn) bfr[tn] = load_bt(Xt + (size_t)(p0 + tn * 16) * CCH + kk, CCH);
#pragma unroll
    for (int tm = 0; tm < 4; ++tm)
#pragma unroll
      for (int tn = 0; tn < 4; ++tn) acc[tm][tn] = wmma_bf16(af[tm], bfr[tn], acc[tm][tn]);
  }

  if (z == 1) {        // Q: pixel-major packed b128 stores
#pragma unroll
    for (int tm = 0; tm < 4; ++tm)
#pragma unroll
      for (int tn = 0; tn < 4; ++tn) {
        int p = p0 + tn * 16 + j;
        int cb = m0 + tm * 16 + h * 8;
        bf16x8 o;
#pragma unroll
        for (int r = 0; r < 8; ++r) o[r] = (bf16_t)acc[tm][tn][r];
        *(bf16x8*)(Qt + (size_t)p * CCH + cb) = o;
      }
  } else {             // K / V: channel-major [N][C][d]
    bf16_t* out = (z == 0) ? Kcm : Vcm;
#pragma unroll
    for (int tm = 0; tm < 4; ++tm)
#pragma unroll
      for (int tn = 0; tn < 4; ++tn) {
        int p = p0 + tn * 16 + j;
        int ni = p >> 10, hw = p & 1023;
        int cb = m0 + tm * 16 + h * 8;
        bf16_t* base = out + (size_t)ni * CCH * DPIX + hw;
#pragma unroll
        for (int r = 0; r < 8; ++r) base[(size_t)(cb + r) * DPIX] = (bf16_t)acc[tm][tn][r];
      }
  }
}

// ---------------- kernel 3: per (n,h): S = V*K^T (64x64, K=1024); O = S*Q (64x1024, K=64) ----------------
__global__ __launch_bounds__(256) void attn(const bf16_t* __restrict__ Kcm,
                                            const bf16_t* __restrict__ Vcm,
                                            const bf16_t* __restrict__ Qt,
                                            bf16_t* __restrict__ Ot) {
  __shared__ bf16_t S[64 * 64];
  int nh = blockIdx.x;
  int ni = nh >> 3, hd = nh & 7;
  int wave = threadIdx.x >> 5;
  int lane = threadIdx.x & 31, j = lane & 15, hf = lane >> 4;
  const bf16_t* Vh = Vcm + (size_t)ni * CCH * DPIX + (size_t)(hd * HDIM) * DPIX;
  const bf16_t* Kh = Kcm + (size_t)ni * CCH * DPIX + (size_t)(hd * HDIM) * DPIX;

  // Phase A: S[c][c'] = sum_D Vh[c][D] * Kh[c'][D]; wave -> (tm, tn0..tn0+1)
  {
    int tm = wave & 3, tn0 = (wave >> 2) * 2;
    v8f a0 = {}, a1 = {};
    for (int kk = 0; kk < DPIX; kk += 32) {
      v16bf af = load_a(Vh + (size_t)(tm * 16) * DPIX + kk, DPIX);
      v16bf b0 = load_bt(Kh + (size_t)(tn0 * 16) * DPIX + kk, DPIX);
      v16bf b1 = load_bt(Kh + (size_t)((tn0 + 1) * 16) * DPIX + kk, DPIX);
      a0 = wmma_bf16(af, b0, a0);
      a1 = wmma_bf16(af, b1, a1);
    }
#pragma unroll
    for (int r = 0; r < 8; ++r) {
      S[(tm * 16 + hf * 8 + r) * 64 + tn0 * 16 + j]       = (bf16_t)a0[r];
      S[(tm * 16 + hf * 8 + r) * 64 + (tn0 + 1) * 16 + j] = (bf16_t)a1[r];
    }
  }
  __syncthreads();

  // Phase B: O[c][p] = sum_c' S[c][c'] * Qh[c'][p]; wave covers 128 pixels
  v16bf saf[4][2];
#pragma unroll
  for (int tm = 0; tm < 4; ++tm)
#pragma unroll
    for (int ks = 0; ks < 2; ++ks) saf[tm][ks] = load_a(&S[tm * 16 * 64 + ks * 32], 64);

  for (int nt = 0; nt < 8; ++nt) {
    int pb = ni * DPIX + wave * 128 + nt * 16;     // global pixel base
    v16bf b0 = load_bt(Qt + (size_t)pb * CCH + hd * HDIM, CCH);
    v16bf b1 = load_bt(Qt + (size_t)pb * CCH + hd * HDIM + 32, CCH);
#pragma unroll
    for (int tm = 0; tm < 4; ++tm) {
      v8f acc = {};
      acc = wmma_bf16(saf[tm][0], b0, acc);
      acc = wmma_bf16(saf[tm][1], b1, acc);
      int p = pb + j;
      int cb = hd * HDIM + tm * 16 + hf * 8;
      bf16x8 o;
#pragma unroll
      for (int r = 0; r < 8; ++r) o[r] = (bf16_t)acc[r];
      *(bf16x8*)(Ot + (size_t)p * CCH + cb) = o;
    }
  }
}

// ---------------- kernel 4: Y = Wo*O + bo (f32 ch-major) + group sum/sumsq atomics ----------------
__global__ __launch_bounds__(32) void out_gemm(const bf16_t* __restrict__ Wob, const float* bo,
                                               const bf16_t* __restrict__ Ot,
                                               float* __restrict__ Y, float* __restrict__ stats) {
  int m0 = blockIdx.y * 64, p0 = blockIdx.x * 64;
  int lane = threadIdx.x & 31, j = lane & 15, h = lane >> 4;

  v8f acc[4][4];
#pragma unroll
  for (int tm = 0; tm < 4; ++tm) {
    v8f ini;
#pragma unroll
    for (int r = 0; r < 8; ++r) ini[r] = bo[m0 + tm * 16 + h * 8 + r];
#pragma unroll
    for (int tn = 0; tn < 4; ++tn) acc[tm][tn] = ini;
  }

  for (int kk = 0; kk < CCH; kk += 32) {
    v16bf af[4], bfr[4];
#pragma unroll
    for (int tm = 0; tm < 4; ++tm) af[tm] = load_a(Wob + (size_t)(m0 + tm * 16) * CCH + kk, CCH);
#pragma unroll
    for (int tn = 0; tn < 4; ++tn) bfr[tn] = load_bt(Ot + (size_t)(p0 + tn * 16) * CCH + kk, CCH);
#pragma unroll
    for (int tm = 0; tm < 4; ++tm)
#pragma unroll
      for (int tn = 0; tn < 4; ++tn) acc[tm][tn] = wmma_bf16(af[tm], bfr[tn], acc[tm][tn]);
  }

  int ni = p0 >> 10;
  float* ssum = stats;
  float* ssq  = stats + NB * GRP;
#pragma unroll
  for (int tm = 0; tm < 4; ++tm) {
    float s[8], q[8];
#pragma unroll
    for (int r = 0; r < 8; ++r) { s[r] = 0.f; q[r] = 0.f; }
    int cb = m0 + tm * 16 + h * 8;
#pragma unroll
    for (int tn = 0; tn < 4; ++tn) {
      int p = p0 + tn * 16 + j;
      int hw = p & 1023;
      float* base = Y + (size_t)ni * CCH * DPIX + hw;
#pragma unroll
      for (int r = 0; r < 8; ++r) {
        float v = acc[tm][tn][r];
        base[(size_t)(cb + r) * DPIX] = v;
        s[r] += v; q[r] += v * v;
      }
    }
#pragma unroll
    for (int msk = 1; msk < 16; msk <<= 1)
#pragma unroll
      for (int r = 0; r < 8; ++r) {
        s[r] += __shfl_xor(s[r], msk, 32);
        q[r] += __shfl_xor(q[r], msk, 32);
      }
    if (j == 0) {
      int g0 = cb >> 2;   // 4 channels per group
      atomicAdd(&ssum[ni * GRP + g0],     s[0] + s[1] + s[2] + s[3]);
      atomicAdd(&ssum[ni * GRP + g0 + 1], s[4] + s[5] + s[6] + s[7]);
      atomicAdd(&ssq [ni * GRP + g0],     q[0] + q[1] + q[2] + q[3]);
      atomicAdd(&ssq [ni * GRP + g0 + 1], q[4] + q[5] + q[6] + q[7]);
    }
  }
}

// ---------------- kernel 5: GroupNorm + affine + residual ----------------
__global__ void finalize(const float* __restrict__ Y, const float* __restrict__ stats,
                         const float* __restrict__ X, const float* __restrict__ gw,
                         const float* __restrict__ gb, float* __restrict__ out) {
  size_t tid = (size_t)blockIdx.x * blockDim.x + threadIdx.x;
  int ni = (int)(tid / ((size_t)CCH * DPIX));
  int rem = (int)(tid % ((size_t)CCH * DPIX));
  int c = rem / DPIX;
  int g = c >> 2;
  const float invn = 1.0f / 4096.0f;   // 4 channels * 1024 pixels per group
  float mean = stats[ni * GRP + g] * invn;
  float ms   = stats[NB * GRP + ni * GRP + g] * invn;
  float var  = ms - mean * mean;
  float inv  = rsqrtf(var + EPSV);
  out[tid] = (Y[tid] - mean) * inv * gw[c] + gb[c] + X[tid];
}

extern "C" void kernel_launch(void* const* d_in, const int* in_sizes, int n_in,
                              void* d_out, int out_size, void* d_ws, size_t ws_size,
                              hipStream_t stream) {
  const float* X  = (const float*)d_in[0];
  const float* Wq = (const float*)d_in[1];
  const float* bq = (const float*)d_in[2];
  const float* Wk = (const float*)d_in[3];
  const float* bk = (const float*)d_in[4];
  const float* Wv = (const float*)d_in[5];
  const float* bv = (const float*)d_in[6];
  const float* Wo = (const float*)d_in[7];
  const float* bo = (const float*)d_in[8];
  const float* gw = (const float*)d_in[9];
  const float* gb = (const float*)d_in[10];
  float* out = (float*)d_out;

  char* w = (char*)d_ws;
  size_t off = 0;
  bf16_t* Wb  = (bf16_t*)(w + off); off += (size_t)4 * CCH * CCH * sizeof(bf16_t);
  bf16_t* Xt  = (bf16_t*)(w + off); off += (size_t)PIX * CCH * sizeof(bf16_t);
  bf16_t* Qt  = (bf16_t*)(w + off); off += (size_t)PIX * CCH * sizeof(bf16_t);
  bf16_t* Kcm = (bf16_t*)(w + off); off += (size_t)PIX * CCH * sizeof(bf16_t);
  bf16_t* Vcm = (bf16_t*)(w + off); off += (size_t)PIX * CCH * sizeof(bf16_t);
  bf16_t* Ot  = (bf16_t*)(w + off); off += (size_t)PIX * CCH * sizeof(bf16_t);
  float*  Y   = (float*) (w + off); off += (size_t)PIX * CCH * sizeof(float);
  float*  st  = (float*) (w + off); off += (size_t)2 * NB * GRP * sizeof(float);

  pack_weights<<<4096, 256, 0, stream>>>(Wq, Wk, Wv, Wo, Wb, st);
  transpose_x<<<dim3(DPIX / 16, CCH / 16, NB), dim3(16, 16), 0, stream>>>(X, Xt);
  qkv_gemm<<<dim3(PIX / 64, CCH / 64, 3), 32, 0, stream>>>(Wb, bq, bk, bv, Xt, Kcm, Qt, Vcm);
  attn<<<NB * NHD, 256, 0, stream>>>(Kcm, Vcm, Qt, Ot);
  out_gemm<<<dim3(PIX / 64, CCH / 64), 32, 0, stream>>>(Wb + (size_t)3 * CCH * CCH, bo, Ot, Y, st);
  finalize<<<(PIX * CCH) / 256, 256, 0, stream>>>(Y, st, X, gw, gb, out);
}
